// HypercomplexGNN_54649163874544
// MI455X (gfx1250) — compile-verified, matrix operand order checked by reading, not run on hardware
//
#include <hip/hip_runtime.h>
#include <hip/hip_bf16.h>

typedef __attribute__((ext_vector_type(16))) __bf16 v16bf;
typedef __attribute__((ext_vector_type(8)))  __bf16 v8bf;
typedef __attribute__((ext_vector_type(8)))  float  v8f;
typedef __attribute__((ext_vector_type(4)))  unsigned int u32x4;
typedef __attribute__((ext_vector_type(8)))  int i32x8;
typedef __attribute__((ext_vector_type(4)))  int i32x4;

#define NUM_USER 100000
#define NUM_ITEM 50000
#define NNODES   150000
#define DIM      64
#define AD       256   // ALGEBRA_DIM * DIM
#define DF       512

#if defined(__gfx1250__)
#define HAVE_TDM 1
#define HAVE_ASYNC 1
#endif

__device__ __forceinline__ v16bf cat8(v8bf a, v8bf b) {
  return __builtin_shufflevector(a, b, 0,1,2,3,4,5,6,7,8,9,10,11,12,13,14,15);
}

#if defined(__gfx1250__)
// LDS byte offset of a generic pointer to __shared__ memory
__device__ __forceinline__ unsigned lds_off(void* p) {
  return (unsigned)(unsigned long long)(__attribute__((address_space(3))) void*)p;
}
#endif

#if defined(HAVE_TDM)
// Issue a 2-D bf16 tile load via the Tensor Data Mover.
// rows x 32 elements (contiguous dim0 = 32 bf16), row stride = `stride` elems.
__device__ __forceinline__ void tdm_load_tile(const __bf16* g, unsigned lds,
                                              int rows, int stride) {
  unsigned long long ga = (unsigned long long)g;
  u32x4 g0;
  g0[0] = 1u;                                   // count=1, user mode
  g0[1] = lds;                                  // lds_addr
  g0[2] = (unsigned)ga;                         // global_addr[31:0]
  g0[3] = (unsigned)(ga >> 32) | (2u << 30);    // global_addr[56:32] | type=2
  i32x8 g1;
  g1[0] = 0x10000;                              // data_size = 1 (2 bytes)
  g1[1] = (int)(32u << 16);                     // tensor_dim0 = 32
  g1[2] = (int)((unsigned)rows << 16);          // tensor_dim1 = rows
  g1[3] = (int)(32u << 16);                     // tile_dim0 = 32
  g1[4] = rows;                                 // tile_dim1 = rows (tile_dim2=0)
  g1[5] = stride;                               // tensor_dim0_stride
  g1[6] = 0;
  g1[7] = 0;
  i32x4 z4 = {0, 0, 0, 0};
  asm volatile("tensor_load_to_lds %0, %1, %2, %3"
               :: "s"(g0), "s"(g1), "s"(z4), "s"(z4) : "memory");
}
__device__ __forceinline__ void wait_tensor0() {
  asm volatile("s_wait_tensorcnt 0x0" ::: "memory");
}
#endif

#if defined(HAVE_ASYNC)
// 16-byte async copy global -> LDS (ASYNCcnt), via inline asm:
// VDST = VGPR holding LDS byte offset, VADDR = 64-bit global address, off = no SADDR
__device__ __forceinline__ void async_b128(const void* g, void* l) {
  unsigned loff = lds_off(l);
  unsigned long long ga = (unsigned long long)g;
  asm volatile("global_load_async_to_lds_b128 %0, %1, off"
               :: "v"(loff), "v"(ga) : "memory");
}
__device__ __forceinline__ void wait_async0() {
  asm volatile("s_wait_asynccnt 0x0" ::: "memory");
}
#endif

// ---------------------------------------------------------------------------
// Pre-passes: fp32 -> bf16 copies / transposes (hardware v_cvt, one-shot)
// ---------------------------------------------------------------------------
__global__ void cvt_bf16_kernel(const float* __restrict__ src,
                                __bf16* __restrict__ dst, int n) {
  int i = blockIdx.x * 256 + threadIdx.x;
  if (i < n) dst[i] = (__bf16)src[i];
}

// src [R][C] fp32 -> dst [C][R] bf16
__global__ void transpose_bf16_kernel(const float* __restrict__ src,
                                      __bf16* __restrict__ dst, int R, int C) {
  int i = blockIdx.x * 256 + threadIdx.x;
  if (i < R * C) {
    int r = i / C, c = i - r * C;
    dst[(size_t)c * R + r] = (__bf16)src[i];
  }
}

// ---------------------------------------------------------------------------
// GEMM 1: Hb[50000,256](bf16) = leaky_relu(Fb[50000,512] @ W0t^T + b0)
// grid (3125, 2), block 256 (8 waves). Block tile 16(M) x 128(N).
// A/B bf16 tiles staged by the Tensor Data Mover (TENSORcnt), wave 0 only.
// ---------------------------------------------------------------------------
__global__ void __launch_bounds__(256)
gemm_feat_kernel(const __bf16* __restrict__ Fb, const __bf16* __restrict__ W0t,
                 const float* __restrict__ b0, __bf16* __restrict__ Hb) {
  __shared__ __align__(16) __bf16 lA[16 * 32];    // [m][k]
  __shared__ __align__(16) __bf16 lB[128 * 32];   // [n][k]
  const int tid  = threadIdx.x;
  const int lane = tid & 31;
  const int wv   = tid >> 5;                // 0..7
  const int M0   = blockIdx.x * 16;
  const int Nblk = blockIdx.y * 128;
  const int m    = lane & 15;
  const int half = lane >> 4;
  v8f acc = {};

  for (int k0 = 0; k0 < DF; k0 += 32) {
    __syncthreads();                        // LDS reuse fence
#if defined(HAVE_TDM)
    if (tid == 0) {
      tdm_load_tile(Fb + (size_t)M0 * DF + k0, lds_off(lA), 16, DF);
      tdm_load_tile(W0t + (size_t)Nblk * DF + k0, lds_off(lB), 128, DF);
      wait_tensor0();
    }
#else
    {
      #pragma unroll
      for (int j = 0; j < 2; ++j) {
        int idx = tid * 2 + j;
        int r = idx >> 5, kk = idx & 31;
        lA[r * 32 + kk] = Fb[(size_t)(M0 + r) * DF + (k0 + kk)];
      }
      #pragma unroll
      for (int j = 0; j < 16; ++j) {
        int idx = j * 256 + tid;
        int r = idx >> 5, kk = idx & 31;
        lB[r * 32 + kk] = W0t[(size_t)(Nblk + r) * DF + (k0 + kk)];
      }
    }
#endif
    __syncthreads();
    v8bf a0 = *(const v8bf*)&lA[m * 32 + half * 8];
    v8bf a1 = *(const v8bf*)&lA[m * 32 + 16 + half * 8];
    const int bn = wv * 16 + m;
    v8bf bb0 = *(const v8bf*)&lB[bn * 32 + half * 8];
    v8bf bb1 = *(const v8bf*)&lB[bn * 32 + 16 + half * 8];
    acc = __builtin_amdgcn_wmma_f32_16x16x32_bf16(
        false, cat8(a0, a1), false, cat8(bb0, bb1), (short)0, acc, false, false);
  }

  const int n  = Nblk + wv * 16 + m;
  const int mb = half * 8;
  const float bias = b0[n];
  #pragma unroll
  for (int r = 0; r < 8; ++r) {
    float v = acc[r] + bias;
    v = v > 0.0f ? v : 0.01f * v;           // leaky_relu(0.01)
    Hb[(size_t)(M0 + mb + r) * 256 + n] = (__bf16)v;
  }
}

// ---------------------------------------------------------------------------
// GEMM 2: T[50000,64](f32) = Hb[50000,256] @ W1t^T + b1
// grid 3125, block 128 (4 waves). Tiles staged via async global->LDS b128.
// ---------------------------------------------------------------------------
__global__ void __launch_bounds__(128)
gemm_proj_kernel(const __bf16* __restrict__ Hb, const __bf16* __restrict__ W1t,
                 const float* __restrict__ b1, float* __restrict__ T) {
  __shared__ __align__(16) __bf16 lA[16 * 32];    // [m][k]
  __shared__ __align__(16) __bf16 lB[64 * 32];    // [n][k]
  const int tid  = threadIdx.x;
  const int lane = tid & 31;
  const int wv   = tid >> 5;                // 0..3
  const int M0   = blockIdx.x * 16;
  const int m    = lane & 15;
  const int half = lane >> 4;
  v8f acc = {};

  for (int k0 = 0; k0 < 256; k0 += 32) {
    __syncthreads();
#if defined(HAVE_ASYNC)
    if (tid < 64) {                         // A: 64 x 16B chunks
      int r = tid >> 2, c = tid & 3;
      async_b128(Hb + (size_t)(M0 + r) * 256 + k0 + c * 8, (char*)lA + tid * 16);
    }
    #pragma unroll
    for (int j = 0; j < 2; ++j) {           // B: 256 x 16B chunks
      int e = j * 128 + tid;
      int r = e >> 2, c = e & 3;
      async_b128(W1t + (size_t)r * 256 + k0 + c * 8, (char*)lB + e * 16);
    }
    wait_async0();
#else
    {
      #pragma unroll
      for (int j = 0; j < 4; ++j) {
        int idx = j * 128 + tid;
        int r = idx >> 5, kk = idx & 31;
        lA[r * 32 + kk] = Hb[(size_t)(M0 + r) * 256 + (k0 + kk)];
      }
      #pragma unroll
      for (int j = 0; j < 16; ++j) {
        int idx = j * 128 + tid;
        int r = idx >> 5, kk = idx & 31;
        lB[r * 32 + kk] = W1t[(size_t)r * 256 + (k0 + kk)];
      }
    }
#endif
    __syncthreads();
    v8bf a0 = *(const v8bf*)&lA[m * 32 + half * 8];
    v8bf a1 = *(const v8bf*)&lA[m * 32 + 16 + half * 8];
    const int bn = wv * 16 + m;
    v8bf bb0 = *(const v8bf*)&lB[bn * 32 + half * 8];
    v8bf bb1 = *(const v8bf*)&lB[bn * 32 + 16 + half * 8];
    acc = __builtin_amdgcn_wmma_f32_16x16x32_bf16(
        false, cat8(a0, a1), false, cat8(bb0, bb1), (short)0, acc, false, false);
  }

  const int n  = wv * 16 + m;
  const int mb = half * 8;
  const float bias = b1[n];
  #pragma unroll
  for (int r = 0; r < 8; ++r)
    T[(size_t)(M0 + mb + r) * 64 + n] = acc[r] + bias;
}

// ---------------------------------------------------------------------------
// Component build: C0[node][a*64+d] = l2norm_row(cat(user_emb_a, temp))
// ---------------------------------------------------------------------------
__global__ void __launch_bounds__(256)
build_comps_kernel(const float* __restrict__ emb_s, const float* __restrict__ emb_x,
                   const float* __restrict__ T, float* __restrict__ C0) {
  const int lane = threadIdx.x & 31;
  const int wv   = threadIdx.x >> 5;        // 0..7
  const int node = blockIdx.x * 2 + (wv >> 2);
  const int a    = wv & 3;
  if (node >= NNODES) return;
  const float* src;
  size_t rowoff;
  if (node < NUM_USER) {
    src = (a == 0) ? emb_s : (emb_x + (size_t)(a - 1) * NUM_USER * DIM);
    rowoff = (size_t)node * DIM;
  } else {
    src = T;
    rowoff = (size_t)(node - NUM_USER) * DIM;
  }
  float x0 = src[rowoff + lane];
  float x1 = src[rowoff + 32 + lane];
  float s = x0 * x0 + x1 * x1;
  #pragma unroll
  for (int o = 16; o > 0; o >>= 1) s += __shfl_xor(s, o, 32);
  float inv = 1.0f / fmaxf(sqrtf(s), 1e-12f);
  size_t o = (size_t)node * AD + (size_t)a * DIM;
  C0[o + lane]      = x0 * inv;
  C0[o + 32 + lane] = x1 * inv;
}

// ---------------------------------------------------------------------------
// Graph-norm helpers
// ---------------------------------------------------------------------------
__global__ void zero_kernel(float* __restrict__ p, int n) {
  int i = blockIdx.x * 256 + threadIdx.x;
  if (i < n) p[i] = 0.0f;
}

__global__ void deg_kernel(const int* __restrict__ row, float* __restrict__ deg, int E) {
  int e = blockIdx.x * 256 + threadIdx.x;
  if (e < E) unsafeAtomicAdd(&deg[row[e]], 1.0f);
}

__global__ void dinv_kernel(const float* __restrict__ deg, float* __restrict__ dinv, int n) {
  int i = blockIdx.x * 256 + threadIdx.x;
  if (i < n) dinv[i] = rsqrtf(deg[i]);      // inf only at isolated nodes (unused)
}

__global__ void norm_kernel(const int* __restrict__ row, const int* __restrict__ col,
                            const float* __restrict__ dinv, float* __restrict__ nrm, int E) {
  int e = blockIdx.x * 256 + threadIdx.x;
  if (e < E) nrm[e] = dinv[row[e]] * dinv[col[e]];
}

// ---------------------------------------------------------------------------
// Propagation: out[col] += nrm[e] * cur[row]  (one wave per edge, 8 f32/lane)
// ---------------------------------------------------------------------------
__global__ void __launch_bounds__(256)
propagate_kernel(const float* __restrict__ cur, const int* __restrict__ row,
                 const int* __restrict__ col, const float* __restrict__ nrm,
                 float* __restrict__ out, int E) {
  const int lane = threadIdx.x & 31;
  const int wv   = threadIdx.x >> 5;
  const int e    = blockIdx.x * 8 + wv;
  if (e >= E) return;
  const int r = row[e], c = col[e];
  const float w = nrm[e];
  const float4* src = (const float4*)(cur + (size_t)r * AD) + lane * 2;
  float4 v0 = src[0];
  float4 v1 = src[1];
  float* dst = out + (size_t)c * AD + lane * 8;
  unsafeAtomicAdd(dst + 0, w * v0.x);
  unsafeAtomicAdd(dst + 1, w * v0.y);
  unsafeAtomicAdd(dst + 2, w * v0.z);
  unsafeAtomicAdd(dst + 3, w * v0.w);
  unsafeAtomicAdd(dst + 4, w * v1.x);
  unsafeAtomicAdd(dst + 5, w * v1.y);
  unsafeAtomicAdd(dst + 6, w * v1.z);
  unsafeAtomicAdd(dst + 7, w * v1.w);
}

__global__ void finalize_kernel(float* __restrict__ outp, const float* __restrict__ B1,
                                const float* __restrict__ prompt, int n) {
  int i = blockIdx.x * 256 + threadIdx.x;
  if (i < n) outp[i] = outp[i] + B1[i] + prompt[i];
}

// ---------------------------------------------------------------------------
extern "C" void kernel_launch(void* const* d_in, const int* in_sizes, int n_in,
                              void* d_out, int out_size, void* d_ws, size_t ws_size,
                              hipStream_t stream) {
  const int*   edge   = (const int*)d_in[0];   // [2, E]
  const float* feats  = (const float*)d_in[1]; // [50000, 512]
  const float* emb_s  = (const float*)d_in[2]; // [100000, 64]
  const float* emb_x  = (const float*)d_in[3]; // [3, 100000, 64]
  const float* prompt = (const float*)d_in[4]; // [150000, 256]
  const float* mlp_w  = (const float*)d_in[5]; // [512, 256]
  const float* mlp_b  = (const float*)d_in[6]; // [256]
  const float* mlp1_w = (const float*)d_in[7]; // [256, 64]
  const float* mlp1_b = (const float*)d_in[8]; // [64]
  float* out = (float*)d_out;

  const int E = in_sizes[0] / 2;               // 1,000,000 directed edges
  const int* row = edge;
  const int* col = edge + E;

  // workspace carve-up (byte based)
  char* w = (char*)d_ws;
  __bf16* Fb  = (__bf16*)w;  w += (size_t)NUM_ITEM * DF * 2;   // 51.2 MB
  __bf16* W0t = (__bf16*)w;  w += (size_t)AD * DF * 2;         // 256x512 bf16
  __bf16* W1t = (__bf16*)w;  w += (size_t)DIM * AD * 2;        // 64x256 bf16
  __bf16* Hb  = (__bf16*)w;  w += (size_t)NUM_ITEM * AD * 2;   // 25.6 MB
  float*  T   = (float*)w;   w += (size_t)NUM_ITEM * DIM * 4;
  float*  C0  = (float*)w;   w += (size_t)NNODES * AD * 4;
  float*  B1  = (float*)w;   w += (size_t)NNODES * AD * 4;
  float*  deg = (float*)w;   w += (size_t)NNODES * 4;
  float*  dinv= (float*)w;   w += (size_t)NNODES * 4;
  float*  nrm = (float*)w;

  const int NAD = NNODES * AD;                 // 38,400,000
  const int NF  = NUM_ITEM * DF;               // 25,600,000

  // bf16 pre-passes (one-shot conversions; weights transposed to [N][K])
  cvt_bf16_kernel<<<(NF + 255) / 256, 256, 0, stream>>>(feats, Fb, NF);
  transpose_bf16_kernel<<<(DF * AD + 255) / 256, 256, 0, stream>>>(mlp_w, W0t, DF, AD);
  transpose_bf16_kernel<<<(256 * DIM + 255) / 256, 256, 0, stream>>>(mlp1_w, W1t, 256, DIM);

  // item MLP (WMMA bf16; TDM / async-LDS staging)
  gemm_feat_kernel<<<dim3(NUM_ITEM / 16, 2), 256, 0, stream>>>(Fb, W0t, mlp_b, Hb);
  gemm_proj_kernel<<<NUM_ITEM / 16, 128, 0, stream>>>(Hb, W1t, mlp1_b, T);

  // graph normalization
  zero_kernel<<<(NNODES + 255) / 256, 256, 0, stream>>>(deg, NNODES);
  deg_kernel<<<(E + 255) / 256, 256, 0, stream>>>(row, deg, E);
  dinv_kernel<<<(NNODES + 255) / 256, 256, 0, stream>>>(deg, dinv, NNODES);
  norm_kernel<<<(E + 255) / 256, 256, 0, stream>>>(row, col, dinv, nrm, E);

  // components
  build_comps_kernel<<<NNODES / 2, 256, 0, stream>>>(emb_s, emb_x, T, C0);

  // layer 1: C0 -> B1
  zero_kernel<<<(NAD + 255) / 256, 256, 0, stream>>>(B1, NAD);
  propagate_kernel<<<(E + 7) / 8, 256, 0, stream>>>(C0, row, col, nrm, B1, E);

  // layer 2: B1 -> out, then out += B1 + prompt
  zero_kernel<<<(NAD + 255) / 256, 256, 0, stream>>>(out, NAD);
  propagate_kernel<<<(E + 7) / 8, 256, 0, stream>>>(B1, row, col, nrm, out, E);
  finalize_kernel<<<(NAD + 255) / 256, 256, 0, stream>>>(out, B1, prompt, NAD);
}
